// Sent_FixedCATS_Attention_81638738362513
// MI455X (gfx1250) — matrix operand order repaired and checked by reading, not compiled, not run on hardware
//
#include <hip/hip_runtime.h>

// Sizes from the reference
#define EMB   768
#define NATT  256
#define BATCH 256
#define SEQ   256
#define HID   256

#define KCH    32                 // K per bf16 WMMA
#define KSTEPS (EMB / KCH)        // 24
#define ROWU   40                 // ushorts per LDS chunk row: 32 data + 8 pad (80B)
                                  // row stride = 20 dwords -> 16 conflict-free bank groups

typedef __attribute__((ext_vector_type(16))) __bf16   v16bf;
typedef __attribute__((ext_vector_type(8)))  float    v8f;
typedef __attribute__((ext_vector_type(4)))  float    vf4;

union FragU { v16bf v; vf4 f[2]; };   // 32B fragment = 2x b128 loads

__device__ __forceinline__ unsigned pack_bf16x2(float lo, float hi) {
  unsigned a = __float_as_uint(lo);
  unsigned b = __float_as_uint(hi);
  a += 0x7FFFu + ((a >> 16) & 1u);     // RNE
  b += 0x7FFFu + ((b >> 16) & 1u);
  return (a >> 16) | (b & 0xFFFF0000u);
}

// Branch-free tanh: exact at +-inf, ~1e-7 abs error (far below bf16 GEMM noise)
__device__ __forceinline__ float fast_tanh(float x) {
  float e2 = __expf(2.f * x);                         // v_exp_f32 path
  return 1.f - 2.f * __builtin_amdgcn_rcpf(e2 + 1.f); // v_rcp_f32
}

__device__ __forceinline__ float wave_red32(float v) {
  v += __shfl_xor(v, 1, 32);
  v += __shfl_xor(v, 2, 32);
  v += __shfl_xor(v, 4, 32);
  v += __shfl_xor(v, 8, 32);
  v += __shfl_xor(v, 16, 32);
  return v;
}

__device__ __forceinline__ float half_red16(float v) {  // reduce lanes 0-15 / 16-31 separately
  v += __shfl_xor(v, 1, 32);
  v += __shfl_xor(v, 2, 32);
  v += __shfl_xor(v, 4, 32);
  v += __shfl_xor(v, 8, 32);
  return v;
}

// -------- Kernel 1: Wa_p (rows 768..1535 of Wa) -> bf16, transposed [n][c] --------
__global__ __launch_bounds__(256) void wat_kernel(const float* __restrict__ Wa,
                                                  unsigned short* __restrict__ WaT) {
  int c = blockIdx.x;          // 0..767
  int n = threadIdx.x;         // 0..255
  unsigned a = __float_as_uint(Wa[(EMB + c) * NATT + n]);
  a += 0x7FFFu + ((a >> 16) & 1u);
  WaT[n * EMB + c] = (unsigned short)(a >> 16);
}

// -------- Kernel 2: q[b][n] = Xq[b] @ Wa_q  (f32) --------
__global__ __launch_bounds__(256) void qproj_kernel(const float* __restrict__ Xq,
                                                    const float* __restrict__ Wa,
                                                    float* __restrict__ qv) {
  __shared__ float xq[EMB];
  int b = blockIdx.x, t = threadIdx.x;
  for (int i = t; i < EMB; i += 256) xq[i] = Xq[b * EMB + i];
  __syncthreads();
  float acc = 0.f;
  for (int c = 0; c < EMB; ++c) acc += xq[c] * Wa[c * NATT + t];   // coalesced over t
  qv[b * NATT + t] = acc;
}

// -------- Kernel 3: attention (one workgroup per (batch, part)) --------
// Software-pipelined: A chunk prefetched to registers, double-buffered in LDS,
// one barrier per K-step; B fragments read directly from L2-resident WaT.
__global__ __launch_bounds__(512, 1) void attend_kernel(const float* __restrict__ Xp,
                                                        const float* __restrict__ va,
                                                        const unsigned short* __restrict__ WaT,
                                                        const float* __restrict__ qv,
                                                        float* __restrict__ Xpdash) {
  __shared__ unsigned short Ach[2][256 * ROWU];  // double-buffered A chunk [s][k] bf16
  __shared__ float qS[NATT];
  __shared__ float vaS[NATT];
  __shared__ float Ssum[SEQ];
  __shared__ __align__(16) float betaS[SEQ];
  __shared__ float red[17];

  const int tid  = threadIdx.x;
  const int lane = tid & 31;
  const int w    = tid >> 5;     // wave 0..15
  const int wm   = w & 1;        // M block: 128 rows (8 tiles)
  const int wn   = w >> 1;       // N block: 32 cols (2 tiles)
  const int hsel = (lane >= 16);
  const int l15  = lane & 15;

  const int b    = blockIdx.x >> 1;
  const int part = blockIdx.x & 1;
  const long part_off = (long)b * (2 * EMB + 2) * SEQ + (part ? (long)(EMB + 1) * SEQ : 0);
  const float* Xpart = Xp + part_off;                 // (EMB, SEQ), s contiguous
  const float* valid = Xp + part_off + (long)EMB * SEQ;

  if (tid < 256) {
    qS[tid]   = qv[b * NATT + tid];
    vaS[tid]  = va[tid];
    Ssum[tid] = 0.f;
  }

  v8f acc[8][2];
  const v8f vzero = {0.f, 0.f, 0.f, 0.f, 0.f, 0.f, 0.f, 0.f};
#pragma unroll
  for (int i = 0; i < 8; ++i) {
    acc[i][0] = vzero;
    acc[i][1] = vzero;
  }

  const int s   = tid & 255;      // staging row
  const int kp0 = tid >> 8;       // 0..1 staging pair phase

  // Prologue: prefetch A chunk for kc=0 into registers (coalesced over s)
  float pre[16];
#pragma unroll
  for (int it = 0; it < 8; ++it) {
    int k = (kp0 + it * 2) * 2;
    pre[it * 2]     = Xpart[(long)k * SEQ + s];
    pre[it * 2 + 1] = Xpart[(long)(k + 1) * SEQ + s];
  }

  __syncthreads();   // qS/vaS/Ssum ready before epilogue use; also first-buffer safety

  for (int kc = 0; kc < KSTEPS; ++kc) {
    unsigned short* Abuf = &Ach[kc & 1][0];

    // ---- stage prefetched A chunk (f32 -> packed bf16) into LDS buffer kc&1 ----
#pragma unroll
    for (int it = 0; it < 8; ++it) {
      int kp = kp0 + it * 2;
      *(unsigned*)&Abuf[s * ROWU + kp * 2] = pack_bf16x2(pre[it * 2], pre[it * 2 + 1]);
    }
    __syncthreads();   // single barrier per step (alternating buffers make this sufficient)

    // ---- issue global prefetch for step kc+1 (overlaps with WMMA below) ----
    if (kc + 1 < KSTEPS) {
#pragma unroll
      for (int it = 0; it < 8; ++it) {
        int k = (kc + 1) * KCH + (kp0 + it * 2) * 2;
        pre[it * 2]     = Xpart[(long)k * SEQ + s];
        pre[it * 2 + 1] = Xpart[(long)(k + 1) * SEQ + s];
      }
    }

    // ---- B fragments straight from L2-resident WaT ([n][c] bf16) ----
    FragU fb[2];
#pragma unroll
    for (int j = 0; j < 2; ++j) {
      int n = wn * 32 + j * 16 + l15;
      const unsigned short* p = WaT + n * EMB + kc * KCH + hsel * 8;
      fb[j].f[0] = *(const vf4*)p;          // lane<16: K0..7   | lane>=16: K8..15
      fb[j].f[1] = *(const vf4*)(p + 16);   // lane<16: K16..23 | lane>=16: K24..31
    }

    // ---- A fragments from LDS (ISA 16x32 bf16 A layout) ----
    FragU fa[8];
#pragma unroll
    for (int i = 0; i < 8; ++i) {
      int r = wm * 128 + i * 16 + l15;
      const char* p = (const char*)&Abuf[r * ROWU] + (hsel ? 16 : 0);
      fa[i].f[0] = *(const vf4*)p;          // K0..7   / K8..15
      fa[i].f[1] = *(const vf4*)(p + 32);   // K16..23 / K24..31
    }

#pragma unroll
    for (int i = 0; i < 8; ++i) {
      acc[i][0] = __builtin_amdgcn_wmma_f32_16x16x32_bf16(
          false, fa[i].v, false, fb[0].v, (short)0, acc[i][0], false, false);
      acc[i][1] = __builtin_amdgcn_wmma_f32_16x16x32_bf16(
          false, fa[i].v, false, fb[1].v, (short)0, acc[i][1], false, false);
    }
  }

  // --- epilogue: S[s] = sum_n tanh(q[n] + P[s,n]) * va[n] ---
  float qn[2], van[2];
#pragma unroll
  for (int j = 0; j < 2; ++j) {
    int n = wn * 32 + j * 16 + l15;
    qn[j]  = qS[n];
    van[j] = vaS[n];
  }
#pragma unroll
  for (int i = 0; i < 8; ++i) {
    float part_s[8];
#pragma unroll
    for (int r = 0; r < 8; ++r) {
      float sum = fast_tanh(qn[0] + acc[i][0][r]) * van[0]
                + fast_tanh(qn[1] + acc[i][1][r]) * van[1];
      part_s[r] = half_red16(sum);          // reduce the 16 lanes sharing this row
    }
    if (l15 == 0) {
      int srow0 = wm * 128 + i * 16 + (hsel ? 8 : 0);
#pragma unroll
      for (int r = 0; r < 8; ++r) atomicAdd(&Ssum[srow0 + r], part_s[r]);  // ds_add_f32
    }
  }
  __syncthreads();

  // --- masked softmax over seq (valid==0 -> S=0 -> e=1, matching reference) ---
  float e = 0.f;
  if (tid < 256) {
    e = __expf(valid[tid] * Ssum[tid]);
    betaS[tid] = e;
  }
  float se = wave_red32(e);
  if (lane == 0) red[w] = se;
  __syncthreads();
  if (tid == 0) {
    float t = 0.f;
    for (int i = 0; i < 16; ++i) t += red[i];
    red[16] = 1.f / t;
  }
  __syncthreads();
  float inv = red[16];
  if (tid < 256) betaS[tid] *= inv;
  __syncthreads();

  // --- phase 2: Xpdash[part][b][c] = sum_s beta[s] * Xpart[c][s] (b128-vectorized) ---
  for (int c = w; c < EMB; c += 16) {
    const vf4* xr = (const vf4*)(Xpart + (long)c * SEQ);   // 16B aligned (1KB row offsets)
    float sacc = 0.f;
#pragma unroll
    for (int i = 0; i < 2; ++i) {
      int q4 = lane + i * 32;                              // vf4 index: 64 per row
      vf4 xv = xr[q4];
      vf4 bv = *(const vf4*)&betaS[q4 * 4];
      sacc += bv.x * xv.x + bv.y * xv.y + bv.z * xv.z + bv.w * xv.w;
    }
    sacc = wave_red32(sacc);
    if (lane == 0) Xpdash[((long)part * BATCH + b) * EMB + c] = sacc;
  }
}

// -------- Kernel 4: o[b] = relu(cat(Xq, Xp1dash, Xp2dash) @ W1 + b1) @ W2 + b2 --------
__global__ __launch_bounds__(256) void mlp_kernel(const float* __restrict__ Xq,
                                                  const float* __restrict__ Xpdash,
                                                  const float* __restrict__ W1,
                                                  const float* __restrict__ b1,
                                                  const float* __restrict__ W2,
                                                  const float* __restrict__ b2,
                                                  float* __restrict__ out) {
  __shared__ float X[3 * EMB];
  __shared__ float red[8];
  int b = blockIdx.x, t = threadIdx.x;
  for (int i = t; i < EMB; i += 256) {
    X[i]           = Xq[b * EMB + i];
    X[EMB + i]     = Xpdash[(long)b * EMB + i];                  // part 0
    X[2 * EMB + i] = Xpdash[((long)BATCH + b) * EMB + i];        // part 1
  }
  __syncthreads();
  float acc = b1[t];
  for (int k = 0; k < 3 * EMB; ++k) acc += X[k] * W1[k * HID + t];  // coalesced over t
  float h = fmaxf(acc, 0.f) * W2[t];
  h = wave_red32(h);
  if ((t & 31) == 0) red[t >> 5] = h;
  __syncthreads();
  if (t == 0) {
    float sres = 0.f;
    for (int i = 0; i < 8; ++i) sres += red[i];
    out[b] = sres + b2[0];
  }
}

extern "C" void kernel_launch(void* const* d_in, const int* in_sizes, int n_in,
                              void* d_out, int out_size, void* d_ws, size_t ws_size,
                              hipStream_t stream) {
  (void)in_sizes; (void)n_in; (void)out_size; (void)ws_size;
  const float* Xq = (const float*)d_in[0];
  const float* Xp = (const float*)d_in[1];
  const float* Wa = (const float*)d_in[2];
  const float* va = (const float*)d_in[3];
  const float* W1 = (const float*)d_in[4];
  const float* b1 = (const float*)d_in[5];
  const float* W2 = (const float*)d_in[6];
  const float* b2 = (const float*)d_in[7];
  float* out = (float*)d_out;

  // Workspace layout (2,228,224 B total)
  char* ws = (char*)d_ws;
  unsigned short* WaT = (unsigned short*)ws;                     // 256*768*2   = 393,216 B
  float* qv           = (float*)(ws + 393216);                   // 256*256*4   = 262,144 B
  float* Xpdash       = (float*)(ws + 393216 + 262144);          // 2*256*768*4 = 1,572,864 B

  wat_kernel   <<<dim3(EMB),       dim3(NATT), 0, stream>>>(Wa, WaT);
  qproj_kernel <<<dim3(BATCH),     dim3(NATT), 0, stream>>>(Xq, Wa, qv);
  attend_kernel<<<dim3(2 * BATCH), dim3(512),  0, stream>>>(Xp, va, WaT, qv, Xpdash);
  mlp_kernel   <<<dim3(BATCH),     dim3(HID),  0, stream>>>(Xq, Xpdash, W1, b1, W2, b2, out);
}
